// RuleMiner_46703474377023
// MI455X (gfx1250) — matrix operand order repaired and checked by reading, compiled, not run on hardware
//
#include <hip/hip_runtime.h>
#include <hip/hip_bf16.h>
#include <math.h>

// Problem constants (match reference)
#define Rr 3
#define Tt 3
#define Nn 50000
#define OPSn 24
#define En 400000
#define Bn 32
#define MAXDEGn 8
#define QDn 128
#define EDn 128
#define Hn 128
#define VOC 25   // OPS+1 distinct degree values / embedding rows

typedef __attribute__((ext_vector_type(16))) _Float16 v16h;
typedef __attribute__((ext_vector_type(8)))  float    v8f;

__device__ __forceinline__ float sigm(float x) { return 1.0f / (1.0f + __expf(-x)); }

// ---------------- generic zero ----------------
__global__ void k_zero(float* p, int n) {
    int i = blockIdx.x * blockDim.x + threadIdx.x;
    if (i < n) p[i] = 0.0f;
}

// ---------------- query: x-projection (constant over T), biases folded ----------------
// xproj[(r*2+d)*B*512 + b*512 + u] = qemb[queries[b]] . qWih[r,d,u,:] + qbih + qbhh
__global__ void k_qxproj(const int* __restrict__ queries, const float* __restrict__ qemb,
                         const float* __restrict__ qWih, const float* __restrict__ qbih,
                         const float* __restrict__ qbhh, float* __restrict__ xproj) {
    int tid = blockIdx.x * blockDim.x + threadIdx.x;       // over 3*2*32*512
    if (tid >= Rr * 2 * Bn * 512) return;
    int u  = tid & 511;
    int b  = (tid >> 9) & 31;
    int rd = tid >> 14;                                    // r*2+d
    const float* e = qemb + (size_t)queries[b] * QDn;
    const float* w = qWih + ((size_t)rd * 512 + u) * QDn;
    float acc = qbih[rd * 512 + u] + qbhh[rd * 512 + u];
    for (int k = 0; k < QDn; ++k) acc += e[k] * w[k];
    xproj[((size_t)rd * Bn + b) * 512 + u] = acc;
}

// ---------------- query LSTM: one block per (r,d), 3 steps, states in LDS ----------------
// hs[((rd*3 + s)*B + b)*128 + u] = h after s+1 steps
__global__ __launch_bounds__(256) void k_qlstm(const float* __restrict__ xproj,
                                               const float* __restrict__ qWhh,
                                               float* __restrict__ hs) {
    __shared__ float sh[Bn][Hn + 4];
    int rd = blockIdx.x;                      // 0..5
    int tid = threadIdx.x;
    const float* xp = xproj + (size_t)rd * Bn * 512;
    const float* W  = qWhh + (size_t)rd * 512 * Hn;
    for (int i = tid; i < Bn * Hn; i += 256) sh[i >> 7][i & 127] = 0.0f;
    float creg[16];
#pragma unroll
    for (int p = 0; p < 16; ++p) creg[p] = 0.0f;
    __syncthreads();
    int u = tid & 127;
    int bhalf = tid >> 7;                     // thread owns (b = 2p+bhalf, u) pairs
    for (int s = 0; s < Tt; ++s) {
        float hv[16];
#pragma unroll 1
        for (int p = 0; p < 16; ++p) {
            int b = 2 * p + bhalf;
            float gi = xp[b * 512 + u];
            float gf = xp[b * 512 + 128 + u];
            float gg = xp[b * 512 + 256 + u];
            float go = xp[b * 512 + 384 + u];
            for (int k = 0; k < Hn; ++k) {
                float hk = sh[b][k];
                gi += hk * W[(u)       * Hn + k];
                gf += hk * W[(128 + u) * Hn + k];
                gg += hk * W[(256 + u) * Hn + k];
                go += hk * W[(384 + u) * Hn + k];
            }
            float c2 = sigm(gf) * creg[p] + sigm(gi) * tanhf(gg);
            creg[p] = c2;
            hv[p] = sigm(go) * tanhf(c2);
        }
        __syncthreads();
#pragma unroll
        for (int p = 0; p < 16; ++p) {
            int b = 2 * p + bhalf;
            sh[b][u] = hv[p];
            hs[(((size_t)rd * Tt + s) * Bn + b) * Hn + u] = hv[p];
        }
        __syncthreads();
    }
}

// ---------------- query attention: linear(25) + softmax per (r,t,b) ----------------
__global__ void k_qattn(const float* __restrict__ hs, const float* __restrict__ qlinW,
                        const float* __restrict__ qlinb, float* __restrict__ qattn) {
    int tid = threadIdx.x;
    if (tid >= Rr * Tt * Bn) return;
    int b = tid & 31;
    int t = (tid >> 5) % Tt;
    int r = tid / (Tt * Bn);
    const float* hf = hs + (((size_t)(r * 2 + 0) * Tt + t) * Bn + b) * Hn;
    const float* hb = hs + (((size_t)(r * 2 + 1) * Tt + (Tt - 1 - t)) * Bn + b) * Hn;  // ys_b[::-1]
    float lg[VOC];
    float mx = -1e30f;
    for (int o = 0; o < VOC; ++o) {
        float a = qlinb[o];
        const float* w = qlinW + (size_t)o * 256;
        for (int k = 0; k < Hn; ++k) a += hf[k] * w[k] + hb[k] * w[128 + k];
        lg[o] = a;
        mx = fmaxf(mx, a);
    }
    float sum = 0.0f;
    for (int o = 0; o < VOC; ++o) { lg[o] = __expf(lg[o] - mx); sum += lg[o]; }
    float inv = 1.0f / sum;
    float* out = qattn + (((size_t)r * Tt + t) * Bn + b) * VOC;
    for (int o = 0; o < VOC; ++o) out[o] = lg[o] * inv;
}

// ---------------- entity: input-projection lookup table (25 rows), biases folded ----------------
__global__ void k_ptab(const float* __restrict__ eWih, const float* __restrict__ ebih,
                       const float* __restrict__ ebhh, const float* __restrict__ eemb,
                       float* __restrict__ ptab) {
    int tid = blockIdx.x * blockDim.x + threadIdx.x;       // over 2*25*512
    if (tid >= 2 * VOC * 512) return;
    int u = tid & 511;
    int v = (tid >> 9) % VOC;
    int d = tid / (VOC * 512);
    const float* e = eemb + (size_t)v * EDn;
    const float* w = eWih + ((size_t)d * 512 + u) * EDn;
    float acc = ebih[d * 512 + u] + ebhh[d * 512 + u];
    for (int k = 0; k < EDn; ++k) acc += e[k] * w[k];
    ptab[tid] = acc;
}

// ---------------- pack Whh^T into WMMA B fragments, contiguous-K order (f16) ----------------
// B tile nT (16 cols), kT (32 K): lane L -> col n = nT*16+(L&15), K base (L>>4)*16, 16 contiguous K.
// WhhP[((d*32+nT)*4+kT)*32+lane][0..15] = Whh[d][n][kT*32+(L>>4)*16+j]
__global__ void k_whhpack(const float* __restrict__ eWhh, _Float16* __restrict__ WhhP) {
    int tid = blockIdx.x * blockDim.x + threadIdx.x;       // over 2*512*128
    if (tid >= 2 * 512 * Hn) return;
    int j    = tid & 15;
    int lane = (tid >> 4) & 31;
    int kT   = (tid >> 9) & 3;
    int nT   = (tid >> 11) & 31;
    int d    = tid >> 16;
    int n = nT * 16 + (lane & 15);
    int k = kT * 32 + (lane >> 4) * 16 + j;
    WhhP[tid] = (_Float16)eWhh[((size_t)d * 512 + n) * Hn + k];
}

// ---------------- entity BiLSTM step: WMMA GEMM (h @ Whh^T) fused with LSTM cell ----------------
// grid (3125, 2): x = 16-entity tile, y = direction. 8 waves per block; wave w covers n-tiles 4w..4w+3.
__global__ __launch_bounds__(256) void k_entstep(const int* __restrict__ deg,
                                                 const float* __restrict__ ptab,
                                                 const _Float16* __restrict__ WhhP,
                                                 _Float16* __restrict__ h,
                                                 float* __restrict__ c,
                                                 int step) {
    __shared__ _Float16 hA[16][Hn + 8];       // A tile, padded against bank alignment
    __shared__ float gs[16][512 + 4];         // gate accumulators
    const int d    = blockIdx.y;
    const int teff = (d == 0) ? step : (MAXDEGn - 1 - step);   // backward scans reversed sequence
    const int m0   = blockIdx.x * 16;
    const int tid  = threadIdx.x;
    const int wave = tid >> 5, lane = tid & 31;
    const int half = lane >> 4, mrow = lane & 15;

    // stage previous h (f16) for 16 entities into LDS (zeros at step 0)
    for (int i = tid; i < 16 * Hn; i += 256) {
        int mm = i >> 7, kk = i & 127;
        hA[mm][kk] = (step == 0) ? (_Float16)0.0f
                                 : h[((size_t)d * Nn + (m0 + mm)) * Hn + kk];
    }
    __syncthreads();

    // GEMM: 16(M) x 512(N) x 128(K), f16 WMMA with f32 accumulate
    for (int q = 0; q < 4; ++q) {
        int nT = wave * 4 + q;
        v8f acc = {};
#pragma unroll
        for (int kT = 0; kT < 4; ++kT) {
            // A fragment (contiguous K, matching packed B): elems 0..7 -> K=kT*32+16*half+j,
            //                                               elems 8..15 -> K=kT*32+16*half+8+j
            v16h a;
            int kb = kT * 32 + half * 16;
#pragma unroll
            for (int j = 0; j < 8; ++j) {
                a[j]     = hA[mrow][kb + j];
                a[8 + j] = hA[mrow][kb + 8 + j];
            }
            // B fragment: pre-packed, one contiguous 32B load per lane
            v16h b = *(const v16h*)&WhhP[((((size_t)d * 32 + nT) * 4 + kT) * 32 + lane) * 16];
            acc = __builtin_amdgcn_wmma_f32_16x16x32_f16(false, a, false, b,
                                                         (short)0, acc, false, false);
        }
        // C layout: element i -> m = i + 8*half, n = nT*16 + mrow
#pragma unroll
        for (int i = 0; i < 8; ++i) gs[i + 8 * half][nT * 16 + mrow] = acc[i];
    }
    __syncthreads();

    // fused LSTM cell update (gate order i,f,g,o along the 512 axis)
    for (int i = tid; i < 16 * Hn; i += 256) {
        int mm = i >> 7, u = i & 127;
        int n = m0 + mm;
        int dv = deg[(size_t)n * MAXDEGn + teff];          // in [0,25)
        const float* pt = ptab + ((size_t)d * VOC + dv) * 512;
        float gi = gs[mm][u]        + pt[u];
        float gf = gs[mm][128 + u]  + pt[128 + u];
        float gg = gs[mm][256 + u]  + pt[256 + u];
        float go = gs[mm][384 + u]  + pt[384 + u];
        size_t idx = ((size_t)d * Nn + n) * Hn + u;
        float cv = (step == 0) ? 0.0f : c[idx];
        cv = sigm(gf) * cv + sigm(gi) * tanhf(gg);
        c[idx] = cv;
        h[idx] = (_Float16)(sigm(go) * tanhf(cv));
    }
}

// ---------------- entity attention: linear(24) + softmax per entity ----------------
__global__ void k_eattn(const _Float16* __restrict__ h, const float* __restrict__ elinW,
                        const float* __restrict__ elinb, float* __restrict__ eattn) {
    int n = blockIdx.x * blockDim.x + threadIdx.x;
    if (n >= Nn) return;
    const _Float16* hf = h + (size_t)n * Hn;
    const _Float16* hb = h + ((size_t)Nn + n) * Hn;
    float lg[OPSn];
    float mx = -1e30f;
    for (int o = 0; o < OPSn; ++o) {
        const float* w = elinW + (size_t)o * 256;
        float a = elinb[o];
        for (int k = 0; k < Hn; ++k) a += (float)hf[k] * w[k] + (float)hb[k] * w[128 + k];
        lg[o] = a;
        mx = fmaxf(mx, a);
    }
    float sum = 0.0f;
    for (int o = 0; o < OPSn; ++o) { lg[o] = __expf(lg[o] - mx); sum += lg[o]; }
    float inv = 1.0f / sum;
    for (int o = 0; o < OPSn; ++o) eattn[(size_t)n * OPSn + o] = lg[o] * inv;
}

// ---------------- w[e] = entity_attn[t_heads[e], rels[e]] ----------------
__global__ void k_wgather(const int* __restrict__ t_heads, const int* __restrict__ rels,
                          const float* __restrict__ eattn, float* __restrict__ w) {
    int e = blockIdx.x * blockDim.x + threadIdx.x;
    if (e < En) w[e] = eattn[(size_t)t_heads[e] * OPSn + rels[e]];
}

// ---------------- memory init scatter: mem[(heads[b], b)] = 1 ----------------
__global__ void k_memscatter(const int* __restrict__ heads, float* __restrict__ mem) {
    int b = threadIdx.x;
    if (b < Bn) mem[(size_t)heads[b] * Bn + b] = 1.0f;
}

// ---------------- edge propagation: wave per edge, lane = batch row ----------------
__global__ __launch_bounds__(256) void k_edge(const int* __restrict__ rels,
                                              const int* __restrict__ t_heads,
                                              const int* __restrict__ t_tails,
                                              const float* __restrict__ w,
                                              const float* __restrict__ qa_rt,  // (B,25)
                                              const float* __restrict__ mem,    // (N,B)
                                              float* __restrict__ added) {      // (N,B)
    long long tid = (long long)blockIdx.x * blockDim.x + threadIdx.x;
    if (tid >= (long long)En * Bn) return;
    int e = (int)(tid >> 5);
    int b = (int)(tid & 31);
    int rel = rels[e];
    int hd = t_heads[e], tl = t_tails[e];
    float wv = w[e];
    float fwd = qa_rt[b * VOC + rel]               * wv * mem[(size_t)hd * Bn + b];
    float rev = qa_rt[b * VOC + rel + OPSn / 2]    * wv * mem[(size_t)tl * Bn + b];
    atomicAdd(&added[(size_t)tl * Bn + b], fwd);
    atomicAdd(&added[(size_t)hd * Bn + b], rev);
}

// ---------------- added += mem * qa[:, -1]; per-b row sums ----------------
__global__ __launch_bounds__(256) void k_addnorm(float* __restrict__ added,
                                                 const float* __restrict__ mem,
                                                 const float* __restrict__ qa_rt,
                                                 float* __restrict__ sums) {
    __shared__ float lsum[Bn];
    int tid = threadIdx.x;
    if (tid < Bn) lsum[tid] = 0.0f;
    __syncthreads();
    int i = blockIdx.x * 256 + tid;                 // N*B divisible by 256
    int b = i & 31;
    float v = added[i] + mem[i] * qa_rt[b * VOC + (VOC - 1)];
    added[i] = v;
    atomicAdd(&lsum[b], v);
    __syncthreads();
    if (tid < Bn) atomicAdd(&sums[tid], lsum[tid]);
}

__global__ void k_norm(const float* __restrict__ added, const float* __restrict__ sums,
                       float* __restrict__ mem) {
    int i = blockIdx.x * blockDim.x + threadIdx.x;
    if (i < Nn * Bn) mem[i] = added[i] / fmaxf(1e-20f, sums[i & 31]);
}

// ---------------- logits[b,n] += mem[n,b] (coalesced writes) ----------------
__global__ void k_accum(const float* __restrict__ mem, float* __restrict__ logits) {
    int i = blockIdx.x * blockDim.x + threadIdx.x;
    if (i >= Bn * Nn) return;
    int b = i / Nn, n = i % Nn;
    logits[i] += mem[(size_t)n * Bn + b];
}

extern "C" void kernel_launch(void* const* d_in, const int* in_sizes, int n_in,
                              void* d_out, int out_size, void* d_ws, size_t ws_size,
                              hipStream_t stream) {
    const int*   queries   = (const int*)d_in[0];
    const int*   heads     = (const int*)d_in[1];
    const int*   rels      = (const int*)d_in[2];
    const int*   t_heads   = (const int*)d_in[3];
    const int*   t_tails   = (const int*)d_in[4];
    const int*   deg       = (const int*)d_in[5];
    const float* qemb      = (const float*)d_in[6];
    const float* eemb      = (const float*)d_in[7];
    const float* q_Wih     = (const float*)d_in[8];
    const float* q_Whh     = (const float*)d_in[9];
    const float* q_bih     = (const float*)d_in[10];
    const float* q_bhh     = (const float*)d_in[11];
    const float* e_Wih     = (const float*)d_in[12];
    const float* e_Whh     = (const float*)d_in[13];
    const float* e_bih     = (const float*)d_in[14];
    const float* e_bhh     = (const float*)d_in[15];
    const float* q_lin_W   = (const float*)d_in[16];
    const float* q_lin_b   = (const float*)d_in[17];
    const float* e_lin_W   = (const float*)d_in[18];
    const float* e_lin_b   = (const float*)d_in[19];
    float* logits = (float*)d_out;

    // workspace carve-out
    char* ws = (char*)d_ws;
    size_t off = 0;
    auto carve = [&](size_t bytes) -> void* {
        void* p = (void*)(ws + off);
        off = (off + bytes + 255) & ~(size_t)255;
        return p;
    };
    float*     xproj = (float*)carve((size_t)Rr * 2 * Bn * 512 * 4);
    float*     hs    = (float*)carve((size_t)Rr * 2 * Tt * Bn * Hn * 4);
    float*     qattn = (float*)carve((size_t)Rr * Tt * Bn * VOC * 4);
    float*     ptab  = (float*)carve((size_t)2 * VOC * 512 * 4);
    _Float16*  WhhP  = (_Float16*)carve((size_t)2 * 512 * Hn * 2);
    _Float16*  hent  = (_Float16*)carve((size_t)2 * Nn * Hn * 2);
    float*     cent  = (float*)carve((size_t)2 * Nn * Hn * 4);
    float*     eattn = (float*)carve((size_t)Nn * OPSn * 4);
    float*     wbuf  = (float*)carve((size_t)En * 4);
    float*     mem   = (float*)carve((size_t)Nn * Bn * 4);
    float*     added = (float*)carve(((size_t)Nn * Bn + Bn) * 4);
    float*     sums  = added + (size_t)Nn * Bn;

    // zero output accumulator (d_out is poisoned)
    k_zero<<<(Bn * Nn + 255) / 256, 256, 0, stream>>>(logits, Bn * Nn);

    // --- query side (tiny) ---
    k_qxproj<<<(Rr * 2 * Bn * 512 + 255) / 256, 256, 0, stream>>>(queries, qemb, q_Wih, q_bih, q_bhh, xproj);
    k_qlstm<<<Rr * 2, 256, 0, stream>>>(xproj, q_Whh, hs);
    k_qattn<<<1, 320, 0, stream>>>(hs, q_lin_W, q_lin_b, qattn);

    // --- entity BiLSTM: precompute input table + packed weights, then 8 WMMA steps ---
    k_ptab<<<(2 * VOC * 512 + 255) / 256, 256, 0, stream>>>(e_Wih, e_bih, e_bhh, eemb, ptab);
    k_whhpack<<<(2 * 512 * Hn + 255) / 256, 256, 0, stream>>>(e_Whh, WhhP);
    for (int s = 0; s < MAXDEGn; ++s)
        k_entstep<<<dim3(Nn / 16, 2), 256, 0, stream>>>(deg, ptab, WhhP, hent, cent, s);
    k_eattn<<<(Nn + 255) / 256, 256, 0, stream>>>(hent, e_lin_W, e_lin_b, eattn);
    k_wgather<<<(En + 255) / 256, 256, 0, stream>>>(t_heads, rels, eattn, wbuf);

    // --- sparse propagation ---
    for (int r = 0; r < Rr; ++r) {
        k_zero<<<(Nn * Bn + 255) / 256, 256, 0, stream>>>(mem, Nn * Bn);
        k_memscatter<<<1, 32, 0, stream>>>(heads, mem);
        for (int t = 0; t < Tt; ++t) {
            const float* qa_rt = qattn + ((size_t)(r * Tt + t) * Bn * VOC);
            k_zero<<<(Nn * Bn + Bn + 255) / 256, 256, 0, stream>>>(added, Nn * Bn + Bn);
            k_edge<<<(int)(((long long)En * Bn + 255) / 256), 256, 0, stream>>>(
                rels, t_heads, t_tails, wbuf, qa_rt, mem, added);
            k_addnorm<<<Nn * Bn / 256, 256, 0, stream>>>(added, mem, qa_rt, sums);
            k_norm<<<(Nn * Bn + 255) / 256, 256, 0, stream>>>(added, sums, mem);
        }
        k_accum<<<(Bn * Nn + 255) / 256, 256, 0, stream>>>(mem, logits);
    }
    (void)in_sizes; (void)n_in; (void)out_size; (void)ws_size;
}
